// VoxelHashTableDynamicFlow_45483703664711
// MI455X (gfx1250) — matrix operand order repaired and checked by reading, not compiled, not run on hardware
//
#include <hip/hip_runtime.h>
#include <cstddef>

typedef __bf16 v16bf __attribute__((ext_vector_type(16)));
typedef __bf16 v8bf  __attribute__((ext_vector_type(8)));
typedef float  v8f   __attribute__((ext_vector_type(8)));
typedef float  v4f   __attribute__((ext_vector_type(4)));
typedef __bf16 v4bf  __attribute__((ext_vector_type(4)));

#define FEAT    120
#define KPAD    128
#define ROWS    64
#define BLOCK   128      // 4 waves (wave32)
#define XSTRIDE 136      // bf16 elems per sX row: 272B, 16B-aligned, bank-skewed
#define TMASK   0xFFFFF  // TABLE_SIZE = 2^20
#define WMAT    16384    // bf16 elems per fragment-layout weight matrix (32KB)

// Decode chunk id -> (n, kb) of an 8-element B-fragment chunk.
// Layout: 32 tiles (ktile 0..3 x ntile 0..7); per tile lane l holds column
// n = ntile*16 + (l&15), K = ktile*32 + 8*(l>>4) + {0..7, 16..23} as two 16B halves.
__device__ __forceinline__ void decode_chunk(int cc, int& n, int& kb, int& off) {
  int ln    = cc & 31;
  int half  = (cc >> 5) & 1;
  int tile  = cc >> 6;          // 0..31
  int ntile = tile & 7;
  int ktile = tile >> 3;
  n   = ntile * 16 + (ln & 15);
  kb  = ktile * 32 + ((ln >> 4) << 3) + half * 16;
  off = tile * 512 + half * 256 + ln * 8;     // bf16 elems
}

__device__ __forceinline__ v8bf make_chunk(const float* __restrict__ Wp,
                                           int n, int kb) {
  v8bf vv;
  #pragma unroll
  for (int e = 0; e < 8; ++e) {
    int k = kb + e;
    float f = (k < FEAT && n < FEAT) ? Wp[k * FEAT + n] : 0.0f;
    vv[e] = (__bf16)f;
  }
  return vv;
}

// Fallback: in-kernel f32->bf16 transpose staging (used when ws too small)
__device__ __forceinline__ void load_w(const float* __restrict__ Wp,
                                       __bf16* sW, int tid) {
  #pragma unroll 1
  for (int c = tid; c < 2048; c += BLOCK) {
    int n, kb, off;
    decode_chunk(c, n, kb, off);
    *(v8bf*)(sW + off) = make_chunk(Wp, n, kb);
  }
}

// Fast path: bulk copy of pre-transformed fragment-layout weights (ws -> LDS)
__device__ __forceinline__ void copy_w(const __bf16* __restrict__ src,
                                       __bf16* sW, int tid) {
  #pragma unroll
  for (int i = tid; i < 2048; i += BLOCK)
    *(v8bf*)(sW + (size_t)i * 8) = *(const v8bf*)(src + (size_t)i * 8);
}

// A fragment (16x32 bf16): lane l row m = l&15; elems 0..7 = K kb..kb+7,
// elems 8..15 = K kb+16..kb+23, kb = ktile*32 + 8*(l>>4).   (ISA 7.12.2)
__device__ __forceinline__ v16bf load_a(const __bf16* sX, int mtile, int lane,
                                        int ktile) {
  int m  = mtile * 16 + (lane & 15);
  int kb = ktile * 32 + ((lane >> 4) << 3);
  const v8bf* lo = (const v8bf*)(sX + m * XSTRIDE + kb);
  const v8bf* hi = (const v8bf*)(sX + m * XSTRIDE + kb + 16);
  return __builtin_shufflevector(*lo, *hi, 0,1,2,3,4,5,6,7,8,9,10,11,12,13,14,15);
}

// B fragment (32x16 bf16): lane l col n = l&15; elems 0..7 = K 8*(l>>4)+0..7, elems 8..15 = K 8*(l>>4)+16..23.
__device__ __forceinline__ v16bf load_b(const __bf16* sW, int lane, int ktile,
                                        int ntile) {
  int tile = ktile * 8 + ntile;
  const v8bf* lo = (const v8bf*)(sW + tile * 512 + lane * 8);
  const v8bf* hi = (const v8bf*)(sW + tile * 512 + 256 + lane * 8);
  return __builtin_shufflevector(*lo, *hi, 0,1,2,3,4,5,6,7,8,9,10,11,12,13,14,15);
}

// ---- prep kernel: convert 4 weight matrices into fragment-ready bf16 in ws ----
__global__ __launch_bounds__(BLOCK) void
prep_weights(const float* __restrict__ f1W, const float* __restrict__ f2W,
             __bf16* __restrict__ wsW) {
  int c = blockIdx.x * BLOCK + threadIdx.x;   // 0..8191
  if (c >= 4 * 2048) return;
  int mat = c >> 11;
  int cc  = c & 2047;
  const float* Wp = ((mat < 2) ? f1W : f2W) + (2 + (mat & 1)) * FEAT * FEAT;
  int n, kb, off;
  decode_chunk(cc, n, kb, off);
  *(v8bf*)(wsW + (size_t)mat * WMAT + off) = make_chunk(Wp, n, kb);
}

__global__ __launch_bounds__(BLOCK) void
voxel_flow_fused(const float* __restrict__ qp,   const int* __restrict__ qt,
                 const int* __restrict__ table,  const float* __restrict__ statf,
                 const float* __restrict__ dynf, const float* __restrict__ temb,
                 const float* __restrict__ f1W,  const float* __restrict__ f1b,
                 const float* __restrict__ f2W,  const float* __restrict__ f2b,
                 const __bf16* __restrict__ wsW, int use_ws,
                 float* __restrict__ out, int M) {
  __shared__ alignas(16) __bf16 sW[32 * 512];        // 32 KB weight tiles
  __shared__ alignas(16) __bf16 sX[ROWS * XSTRIDE];  // 17 KB activations
  __shared__ float sB[4 * 128];
  __shared__ int   sV[ROWS];
  __shared__ int   sOK[ROWS];
  __shared__ int   sT[ROWS];

  const int tid   = threadIdx.x;
  const int lane  = tid & 31;
  const int mtile = tid >> 5;              // wave id: 4 waves -> 4 M-tiles
  const int base  = blockIdx.x * ROWS;

  // ---- hash + table lookup (one row per thread) ----
  if (tid < ROWS) {
    int r = base + tid; if (r >= M) r = M - 1;
    float px = qp[3 * r + 0], py = qp[3 * r + 1], pz = qp[3 * r + 2];
    int gx = (int)floorf(px / 0.1f);
    int gy = (int)floorf(py / 0.1f);
    int gz = (int)floorf(pz / 0.1f);
    // PRIMES % 2^20: 455773, 475301, 655287; & mask == floor-mod for 2^20
    int h = (gx * 455773 + gy * 475301 + gz * 655287) & TMASK;
    int v = table[h];
    sOK[tid] = (v >= 0);
    sV[tid]  = (v >= 0) ? v : 0;
    int t = qt[r] % 201; if (t < 0) t += 201;
    sT[tid] = t;
  }
  // biases: stage 0..3 -> f1_b[2], f1_b[3], f2_b[2], f2_b[3]
  for (int i = tid; i < 4 * 128; i += BLOCK) {
    int st = i >> 7, c = i & 127;
    const float* bp = ((st < 2) ? f1b : f2b) + (2 + (st & 1)) * FEAT;
    sB[i] = (c < FEAT) ? bp[c] : 0.0f;
  }
  __syncthreads();

  // ---- stage-0 input: gathered time embeddings -> bf16 (vectorized) ----
  #pragma unroll 1
  for (int i = tid; i < ROWS * 30; i += BLOCK) {     // 30 float4 per row
    int r = i / 30, c4 = i - r * 30;
    v4f f = *(const v4f*)(temb + (size_t)sT[r] * FEAT + c4 * 4);
    v4bf b4; b4[0] = (__bf16)f[0]; b4[1] = (__bf16)f[1];
             b4[2] = (__bf16)f[2]; b4[3] = (__bf16)f[3];
    *(v4bf*)(sX + r * XSTRIDE + c4 * 4) = b4;        // 8B store, 8B-aligned
  }
  for (int i = tid; i < ROWS * 8; i += BLOCK) {      // zero pad cols 120..127
    int r = i >> 3, c = 120 + (i & 7);
    sX[r * XSTRIDE + c] = (__bf16)0.0f;
  }

  const float* Wstage[4] = { f1W + 2 * FEAT * FEAT, f1W + 3 * FEAT * FEAT,
                             f2W + 2 * FEAT * FEAT, f2W + 3 * FEAT * FEAT };
  if (use_ws) copy_w(wsW, sW, tid);
  else        load_w(Wstage[0], sW, tid);
  __syncthreads();

  // ---- 4 chained GEMM stages, each 64x128 @ 128x128 via WMMA bf16 ----
  #pragma unroll 1
  for (int stage = 0; stage < 4; ++stage) {
    if (use_ws && stage < 3)   // warm next stage's weights toward L0/L2
      __builtin_prefetch(wsW + (size_t)(stage + 1) * WMAT + tid * 128, 0, 1);

    v8f acc[8] = {};
    #pragma unroll
    for (int kt = 0; kt < 4; ++kt) {
      v16bf a = load_a(sX, mtile, lane, kt);
      #pragma unroll
      for (int nt = 0; nt < 8; ++nt) {
        v16bf b = load_b(sW, lane, kt, nt);
        acc[nt] = __builtin_amdgcn_wmma_f32_16x16x32_bf16(
            false, a, false, b, (short)0, acc[nt], false, false);
      }
    }
    __syncthreads();   // all LDS reads of sX/sW done

    // epilogue: C/D layout -> lane l: n = l&15, m = r + 8*(l>>4)
    #pragma unroll
    for (int nt = 0; nt < 8; ++nt) {
      int n = nt * 16 + (lane & 15);
      #pragma unroll
      for (int r8 = 0; r8 < 8; ++r8) {
        int m = mtile * 16 + r8 + ((lane >> 4) << 3);
        float val = acc[nt][r8] + sB[stage * 128 + n];
        if (stage == 3) {
          if (n < FEAT) {
            int row = base + m;
            float o = sOK[m] ? (val + statf[(size_t)sV[m] * FEAT + n]) : 0.0f;
            if (row < M) out[(size_t)row * FEAT + n] = o;
          }
        } else {
          if (stage == 1 && n < FEAT)
            val += dynf[(size_t)sV[m] * FEAT + n];   // residual: dynamic
          sX[m * XSTRIDE + n] = (__bf16)((n < FEAT) ? val : 0.0f);
        }
      }
    }
    if (stage < 3) {
      if (use_ws) copy_w(wsW + (size_t)(stage + 1) * WMAT, sW, tid);
      else        load_w(Wstage[stage + 1], sW, tid);
    }
    __syncthreads();
  }
}

extern "C" void kernel_launch(void* const* d_in, const int* in_sizes, int n_in,
                              void* d_out, int out_size, void* d_ws, size_t ws_size,
                              hipStream_t stream) {
  const float* qp    = (const float*)d_in[0];
  const int*   qt    = (const int*)  d_in[1];
  const int*   table = (const int*)  d_in[2];
  const float* statf = (const float*)d_in[3];
  const float* dynf  = (const float*)d_in[4];
  const float* temb  = (const float*)d_in[5];
  const float* f1W   = (const float*)d_in[6];
  const float* f1b   = (const float*)d_in[7];
  const float* f2W   = (const float*)d_in[8];
  const float* f2b   = (const float*)d_in[9];
  float* out = (float*)d_out;
  int M = in_sizes[1];

  const size_t ws_needed = (size_t)4 * WMAT * sizeof(__bf16);   // 128 KB
  int use_ws = (ws_size >= ws_needed) ? 1 : 0;
  __bf16* wsW = (__bf16*)d_ws;

  if (use_ws) {
    hipLaunchKernelGGL(prep_weights, dim3((4 * 2048 + BLOCK - 1) / BLOCK),
                       dim3(BLOCK), 0, stream, f1W, f2W, wsW);
  }
  int grid = (M + ROWS - 1) / ROWS;
  hipLaunchKernelGGL(voxel_flow_fused, dim3(grid), dim3(BLOCK), 0, stream,
                     qp, qt, table, statf, dynf, temb, f1W, f1b, f2W, f2b,
                     wsW, use_ws, out, M);
}
